// Attention_523986010701
// MI455X (gfx1250) — matrix operand order repaired and checked by reading, not compiled, not run on hardware
//
#include <hip/hip_runtime.h>

// MI455X (gfx1250, wave32) causal attention layer.
// fp32 in/out; one-shot bf16 conversion prep kernels; all GEMM/attention tiles
// moved with async global->LDS copies (ASYNCcnt, double-buffered); matrix math on
// v_wmma_f32_16x16x32_bf16 with fp32 accumulation; fp32 softmax.

#define NH    16
#define DM    1024
#define DH    64
#define BATCH 4
#define SEQ   2048

typedef __attribute__((ext_vector_type(16))) __bf16         bf16x16;
typedef __attribute__((ext_vector_type(8)))  float          f32x8;
typedef __attribute__((ext_vector_type(8)))  unsigned short u16x8;
typedef __attribute__((ext_vector_type(16))) unsigned short u16x16;

__device__ __forceinline__ unsigned short f2bf(float f) {
  unsigned int u = __builtin_bit_cast(unsigned int, f);
  u += 0x7FFFu + ((u >> 16) & 1u);   // round-to-nearest-even
  return (unsigned short)(u >> 16);
}

__device__ __forceinline__ f32x8 wmma_bf16(u16x16 a, u16x16 b, f32x8 c) {
  return __builtin_amdgcn_wmma_f32_16x16x32_bf16(
      false, __builtin_bit_cast(bf16x16, a),
      false, __builtin_bit_cast(bf16x16, b),
      (short)0, c, false, false);
}

// A fragment (16x32 bf16) from row-major [16][ldk] (LDS or global).
__device__ __forceinline__ u16x16 frag_a(const unsigned short* p0, int ldk, int lane) {
  int m = lane & 15, h = lane >> 4;
  const unsigned short* p = p0 + m * ldk + h * 8;
  u16x8 c0 = *(const u16x8*)(p);
  u16x8 c1 = *(const u16x8*)(p + 16);
  u16x16 r;
#pragma unroll
  for (int i = 0; i < 8; ++i) { r[i] = c0[i]; r[i + 8] = c1[i]; }
  return r;
}

// B fragment (32x16 bf16) from K-major storage [16 cols][ldk].
__device__ __forceinline__ u16x16 frag_b(const unsigned short* p0, int ldk, int lane) {
  int n = lane & 15, h = lane >> 4;
  const unsigned short* p = p0 + n * ldk + h * 8;
  u16x8 c0 = *(const u16x8*)(p);
  u16x8 c1 = *(const u16x8*)(p + 16);
  u16x16 r;
#pragma unroll
  for (int i = 0; i < 8; ++i) { r[i] = c0[i]; r[i + 8] = c1[i]; }
  return r;
}

// CDNA5 async global->LDS copy, 16B per lane (tracked by ASYNCcnt).
__device__ __forceinline__ void async_cp16(unsigned lds_off, const unsigned short* g) {
  asm volatile("global_load_async_to_lds_b128 %0, %1, off"
               :: "v"(lds_off), "v"(g) : "memory");
}
__device__ __forceinline__ void wait_async0() {
  asm volatile("s_wait_asynccnt 0x0" ::: "memory");
}
__device__ __forceinline__ unsigned ldsoff(const void* p) {
  return (unsigned)(size_t)p;   // generic LDS addr: low 32 bits = LDS offset
}

// ---------------- Prep kernels: one-shot fp32 -> bf16 ----------------
__global__ __launch_bounds__(256) void xcvt_kernel(const float* __restrict__ in,
                                                   unsigned short* __restrict__ out) {
  size_t idx = ((size_t)blockIdx.x * 256 + threadIdx.x) * 8;
  u16x8 v;
#pragma unroll
  for (int i = 0; i < 8; ++i) v[i] = f2bf(in[idx + i]);
  *(u16x8*)&out[idx] = v;
}

// Transpose-convert: in [mat][K][N] fp32 -> out [mat][N][K] bf16 (K-major).
__global__ __launch_bounds__(256) void wcvt_kernel(const float* __restrict__ in,
                                                   unsigned short* __restrict__ out,
                                                   int K, int N) {
  size_t mat = blockIdx.z;
  int idx = blockIdx.x * 256 + threadIdx.x;
  if (idx >= N * (K / 8)) return;
  int n = idx / (K / 8);
  int kc = (idx % (K / 8)) * 8;
  const float* src = in + mat * (size_t)K * N;
  u16x8 v;
#pragma unroll
  for (int i = 0; i < 8; ++i) v[i] = f2bf(src[(size_t)(kc + i) * N + n]);
  *(u16x8*)&out[mat * (size_t)N * K + (size_t)n * K + kc] = v;
}

// ---------------- Kernel 1: fused QKV projection ----------------
// grid (64, 16). Inputs all bf16 (x, K-major weights); async double-buffered tiles.
// Q,K out: bf16 [B,H,S,DH]; V out: bf16 [B,H,DH,S] (pre-transposed via LDS).
__device__ __forceinline__ void qkv_issue(const unsigned short* xb_m0,
                                          const unsigned short* wq,
                                          const unsigned short* wk,
                                          const unsigned short* wv,
                                          unsigned ab, unsigned qb, unsigned kbo,
                                          unsigned vb, int kk, int tid) {
#pragma unroll
  for (int rep = 0; rep < 2; ++rep) {             // A tile: 128x32, 512 chunks
    int c = rep * 256 + tid, r = c >> 2, j = c & 3;
    async_cp16(ab + c * 16, xb_m0 + (size_t)r * DM + kk + j * 8);
  }
  int n = tid >> 2, j = tid & 3;                  // B tiles: 64x32, 256 chunks each
  size_t off = (size_t)n * DM + kk + j * 8;
  async_cp16(qb  + tid * 16, wq + off);
  async_cp16(kbo + tid * 16, wk + off);
  async_cp16(vb  + tid * 16, wv + off);
}

__global__ __launch_bounds__(256) void qkv_kernel(
    const unsigned short* __restrict__ xb,
    const unsigned short* __restrict__ WQt, const float* __restrict__ bQ,
    const unsigned short* __restrict__ WKt, const float* __restrict__ bK,
    const unsigned short* __restrict__ WVt, const float* __restrict__ bV,
    unsigned short* __restrict__ Qo, unsigned short* __restrict__ Ko,
    unsigned short* __restrict__ Vo) {
  __shared__ __align__(16) unsigned short As[2][128 * 32];
  __shared__ __align__(16) unsigned short Bq[2][64 * 32];
  __shared__ __align__(16) unsigned short Bk[2][64 * 32];
  __shared__ __align__(16) unsigned short Bv[2][64 * 32];
  __shared__ __align__(16) unsigned short Vts[DH * 128];   // [d][m_local]

  int tid = threadIdx.x, wave = tid >> 5, lane = tid & 31;
  int hh = lane >> 4, n16 = lane & 15;
  int m0 = blockIdx.x * 128;
  int head = blockIdx.y;
  const unsigned short* xm = xb + (size_t)m0 * DM;
  const unsigned short* wq = WQt + (size_t)head * DH * DM; // [64][1024] K-major
  const unsigned short* wk = WKt + (size_t)head * DH * DM;
  const unsigned short* wv = WVt + (size_t)head * DH * DM;

  unsigned ab[2] = { ldsoff(&As[0][0]), ldsoff(&As[1][0]) };
  unsigned qb[2] = { ldsoff(&Bq[0][0]), ldsoff(&Bq[1][0]) };
  unsigned kb[2] = { ldsoff(&Bk[0][0]), ldsoff(&Bk[1][0]) };
  unsigned vb[2] = { ldsoff(&Bv[0][0]), ldsoff(&Bv[1][0]) };

  qkv_issue(xm, wq, wk, wv, ab[0], qb[0], kb[0], vb[0], 0, tid);

  f32x8 aq[4] = {}, ak[4] = {}, av[4] = {};
  for (int step = 0; step < DM / 32; ++step) {
    int kk = step * 32, buf = step & 1;
    wait_async0();
    __syncthreads();
    if (kk + 32 < DM)
      qkv_issue(xm, wq, wk, wv, ab[buf ^ 1], qb[buf ^ 1], kb[buf ^ 1], vb[buf ^ 1],
                kk + 32, tid);
    u16x16 af = frag_a(&As[buf][wave * 16 * 32], 32, lane);
#pragma unroll
    for (int t = 0; t < 4; ++t) {
      aq[t] = wmma_bf16(af, frag_b(&Bq[buf][t * 16 * 32], 32, lane), aq[t]);
      ak[t] = wmma_bf16(af, frag_b(&Bk[buf][t * 16 * 32], 32, lane), ak[t]);
      av[t] = wmma_bf16(af, frag_b(&Bv[buf][t * 16 * 32], 32, lane), av[t]);
    }
  }

  int b = m0 >> 11, s0 = m0 & (SEQ - 1);
  size_t rowbase = ((size_t)(b * NH + head)) * SEQ;
#pragma unroll
  for (int t = 0; t < 4; ++t) {
    int n = t * 16 + n16;
    float bq_ = bQ[head * DH + n], bk_ = bK[head * DH + n], bv_ = bV[head * DH + n];
#pragma unroll
    for (int r = 0; r < 8; ++r) {
      int ml = wave * 16 + r + hh * 8;            // C/D layout: VGPR r -> M=r / M=r+8
      size_t o = (rowbase + s0 + ml) * DH + n;
      Qo[o] = f2bf(aq[t][r] + bq_);
      Ko[o] = f2bf(ak[t][r] + bk_);
      Vts[n * 128 + ml] = f2bf(av[t][r] + bv_);   // transpose V through LDS
    }
  }
  __syncthreads();
  size_t vtbase = ((size_t)(b * NH + head)) * DH * SEQ;
#pragma unroll
  for (int rep = 0; rep < 4; ++rep) {             // coalesced 16B stores of V^T
    int c = rep * 256 + tid;                      // 1024 chunks: 64 rows x 16
    int d = c >> 4, j = c & 15;
    *(u16x8*)&Vo[vtbase + (size_t)d * SEQ + s0 + j * 8] =
        *(const u16x8*)&Vts[d * 128 + j * 8];
  }
}

// ---------------- Kernel 2: causal flash attention ----------------
// grid (S/128, NH, B). Q fragments in registers; K/V^T double-buffered async tiles.
__global__ __launch_bounds__(256) void attn_kernel(
    const unsigned short* __restrict__ Q, const unsigned short* __restrict__ K,
    const unsigned short* __restrict__ Vt, unsigned short* __restrict__ Z) {
  __shared__ __align__(16) unsigned short Ks[2][64 * DH];   // [key][d]
  __shared__ __align__(16) unsigned short Vs[2][64 * DH];   // [d][key]
  __shared__ __align__(16) unsigned short Ps[8 * 16 * 64];  // per-wave P tile

  int tid = threadIdx.x, wave = tid >> 5, lane = tid & 31;
  int hh = lane >> 4, n16 = lane & 15;
  int b = blockIdx.z, head = blockIdx.y;
  int q0 = blockIdx.x * 128;
  const size_t base = ((size_t)(b * NH + head)) * SEQ * DH;
  const unsigned short* Kg = K + base;
  const unsigned short* Vg = Vt + base;

  u16x16 qf[2];
  {
    const unsigned short* qp = Q + base + (size_t)(q0 + wave * 16) * DH;
    qf[0] = frag_a(qp, DH, lane);
    qf[1] = frag_a(qp + 32, DH, lane);
  }

  unsigned ksb[2] = { ldsoff(&Ks[0][0]), ldsoff(&Ks[1][0]) };
  unsigned vsb[2] = { ldsoff(&Vs[0][0]), ldsoff(&Vs[1][0]) };

#pragma unroll
  for (int rep = 0; rep < 2; ++rep) {             // prologue: key block 0 -> buf 0
    int c = rep * 256 + tid;
    async_cp16(ksb[0] + c * 16, Kg + c * 8);
    int d = c >> 4, j = c & 15;
    async_cp16(vsb[0] + c * 16, Vg + (size_t)d * SEQ + j * 8);
  }

  f32x8 accz[4] = {};
  float m_i[8], l_i[8];
#pragma unroll
  for (int r = 0; r < 8; ++r) { m_i[r] = -1e30f; l_i[r] = 0.f; }

  int nkb = (q0 + 128) >> 6;
  for (int kb = 0; kb < nkb; ++kb) {
    int buf = kb & 1;
    wait_async0();
    __syncthreads();
    if (kb + 1 < nkb) {
      int k0n = (kb + 1) * 64;
#pragma unroll
      for (int rep = 0; rep < 2; ++rep) {
        int c = rep * 256 + tid;
        async_cp16(ksb[buf ^ 1] + c * 16, Kg + (size_t)k0n * DH + c * 8);
        int d = c >> 4, j = c & 15;
        async_cp16(vsb[buf ^ 1] + c * 16, Vg + (size_t)d * SEQ + k0n + j * 8);
      }
    }

    int k0 = kb * 64;
    bool part = (k0 <= q0 + wave * 16);           // wave-uniform causal skip
    if (part) {
      const unsigned short* Kt = &Ks[buf][0];
      const unsigned short* Vtt = &Vs[buf][0];
      f32x8 accs[4] = {};
#pragma unroll
      for (int ks = 0; ks < 2; ++ks)
#pragma unroll
        for (int t = 0; t < 4; ++t)
          accs[t] = wmma_bf16(qf[ks], frag_b(Kt + t * 16 * DH + ks * 32, DH, lane),
                              accs[t]);
#pragma unroll
      for (int t = 0; t < 4; ++t) {
        int key = k0 + t * 16 + n16;
#pragma unroll
        for (int r = 0; r < 8; ++r) {
          int qrow = q0 + wave * 16 + r + hh * 8;
          float s = accs[t][r] * 0.125f;          // 1/sqrt(64)
          if (key > qrow) s = -1e30f;             // causal mask
          accs[t][r] = s;
        }
      }
      unsigned short* Pw = Ps + wave * (16 * 64);
#pragma unroll
      for (int r = 0; r < 8; ++r) {               // online softmax per row
        float mx = fmaxf(fmaxf(accs[0][r], accs[1][r]), fmaxf(accs[2][r], accs[3][r]));
#pragma unroll
        for (int off = 1; off < 16; off <<= 1) mx = fmaxf(mx, __shfl_xor(mx, off, 32));
        float mnew = fmaxf(m_i[r], mx);
        float alpha = __expf(m_i[r] - mnew);
        m_i[r] = mnew;
        float rs = 0.f;
#pragma unroll
        for (int t = 0; t < 4; ++t) {
          float p = __expf(accs[t][r] - mnew);
          accs[t][r] = p;
          rs += p;
        }
#pragma unroll
        for (int off = 1; off < 16; off <<= 1) rs += __shfl_xor(rs, off, 32);
        l_i[r] = l_i[r] * alpha + rs;
#pragma unroll
        for (int t = 0; t < 4; ++t) accz[t][r] *= alpha;
      }
#pragma unroll
      for (int t = 0; t < 4; ++t)                 // C-layout -> A-layout via LDS
#pragma unroll
        for (int r = 0; r < 8; ++r)
          Pw[(r + hh * 8) * 64 + t * 16 + n16] = f2bf(accs[t][r]);
#pragma unroll
      for (int ks = 0; ks < 2; ++ks) {            // P @ V
        u16x16 af = frag_a(Pw + ks * 32, 64, lane);
#pragma unroll
        for (int t = 0; t < 4; ++t)
          accz[t] = wmma_bf16(af, frag_b(Vtt + t * 16 * DH + ks * 32, DH, lane),
                              accz[t]);
      }
    }
  }

#pragma unroll
  for (int t = 0; t < 4; ++t) {
#pragma unroll
    for (int r = 0; r < 8; ++r) {
      int qrow = q0 + wave * 16 + r + hh * 8;
      int d = t * 16 + n16;
      Z[((size_t)(b * SEQ + qrow)) * DM + head * DH + d] = f2bf(accz[t][r] / l_i[r]);
    }
  }
}

// ---------------- Kernel 3: output projection ----------------
// out[8192,1024] = z_bf16 @ W_O^T(bf16, K-major) + b_O. grid (64,16). Async tiles.
__device__ __forceinline__ void oproj_issue(const unsigned short* zm,
                                            const unsigned short* wo,
                                            unsigned ab, unsigned bb, int kk, int tid) {
#pragma unroll
  for (int rep = 0; rep < 2; ++rep) {             // A tile: 128x32
    int c = rep * 256 + tid, r = c >> 2, j = c & 3;
    async_cp16(ab + c * 16, zm + (size_t)r * DM + kk + j * 8);
  }
  int n = tid >> 2, j = tid & 3;                  // B tile: 64x32
  async_cp16(bb + tid * 16, wo + (size_t)n * DM + kk + j * 8);
}

__global__ __launch_bounds__(256) void oproj_kernel(
    const unsigned short* __restrict__ Zin, const unsigned short* __restrict__ WOt,
    const float* __restrict__ bO, float* __restrict__ out) {
  __shared__ __align__(16) unsigned short As[2][128 * 32];
  __shared__ __align__(16) unsigned short Bs[2][64 * 32];

  int tid = threadIdx.x, wave = tid >> 5, lane = tid & 31;
  int hh = lane >> 4, n16 = lane & 15;
  int m0 = blockIdx.x * 128, n0 = blockIdx.y * 64;
  const unsigned short* zm = Zin + (size_t)m0 * DM;
  const unsigned short* wo = WOt + (size_t)n0 * DM;  // [1024][1024] K-major

  unsigned ab[2] = { ldsoff(&As[0][0]), ldsoff(&As[1][0]) };
  unsigned bb[2] = { ldsoff(&Bs[0][0]), ldsoff(&Bs[1][0]) };

  oproj_issue(zm, wo, ab[0], bb[0], 0, tid);

  f32x8 acc[4] = {};
  for (int step = 0; step < DM / 32; ++step) {
    int kk = step * 32, buf = step & 1;
    wait_async0();
    __syncthreads();
    if (kk + 32 < DM)
      oproj_issue(zm, wo, ab[buf ^ 1], bb[buf ^ 1], kk + 32, tid);
    u16x16 af = frag_a(&As[buf][wave * 16 * 32], 32, lane);
#pragma unroll
    for (int t = 0; t < 4; ++t)
      acc[t] = wmma_bf16(af, frag_b(&Bs[buf][t * 16 * 32], 32, lane), acc[t]);
  }

#pragma unroll
  for (int t = 0; t < 4; ++t) {
    int n = n0 + t * 16 + n16;
    float bias = bO[n];
#pragma unroll
    for (int r = 0; r < 8; ++r) {
      int row = m0 + wave * 16 + r + hh * 8;
      out[(size_t)row * DM + n] = acc[t][r] + bias;
    }
  }
}

extern "C" void kernel_launch(void* const* d_in, const int* in_sizes, int n_in,
                              void* d_out, int out_size, void* d_ws, size_t ws_size,
                              hipStream_t stream) {
  (void)in_sizes; (void)n_in; (void)out_size; (void)ws_size;
  const float* x  = (const float*)d_in[0];
  // d_in[1] = pos_embed: unused by reference computation (already in residual stream)
  const float* WQ = (const float*)d_in[2];
  const float* bQ = (const float*)d_in[3];
  const float* WK = (const float*)d_in[4];
  const float* bK = (const float*)d_in[5];
  const float* WV = (const float*)d_in[6];
  const float* bV = (const float*)d_in[7];
  const float* WO = (const float*)d_in[8];
  const float* bO = (const float*)d_in[9];
  float* out = (float*)d_out;

  size_t perX = (size_t)BATCH * SEQ * DM;         // 8,388,608 elems
  size_t perW = (size_t)NH * DM * DH;             // 1,048,576 elems
  unsigned short* xb  = (unsigned short*)d_ws;    // bf16 x
  unsigned short* wqt = xb + perX;                // bf16 W_Q^T [H][DH][DM]
  unsigned short* wkt = wqt + perW;
  unsigned short* wvt = wkt + perW;
  unsigned short* wot = wvt + perW;               // bf16 W_O^T [DM][DM]
  unsigned short* q   = wot + (size_t)DM * DM;
  unsigned short* k   = q + perX;
  unsigned short* vt  = k + perX;                 // V stored [B,H,DH,S]
  unsigned short* z   = vt + perX;

  // one-shot conversions
  xcvt_kernel<<<dim3(perX / 8 / 256), 256, 0, stream>>>(x, xb);
  wcvt_kernel<<<dim3((DM * DH / 8) / 256, 1, NH), 256, 0, stream>>>(WQ, wqt, DM, DH);
  wcvt_kernel<<<dim3((DM * DH / 8) / 256, 1, NH), 256, 0, stream>>>(WK, wkt, DM, DH);
  wcvt_kernel<<<dim3((DM * DH / 8) / 256, 1, NH), 256, 0, stream>>>(WV, wvt, DM, DH);
  wcvt_kernel<<<dim3((DM * DM / 8) / 256, 1, 1), 256, 0, stream>>>(WO, wot, DM, DM);

  qkv_kernel<<<dim3((BATCH * SEQ) / 128, NH), 256, 0, stream>>>(
      xb, wqt, bQ, wkt, bK, wvt, bV, q, k, vt);
  attn_kernel<<<dim3(SEQ / 128, NH, BATCH), 256, 0, stream>>>(q, k, vt, z);
  oproj_kernel<<<dim3((BATCH * SEQ) / 128, DM / 64), 256, 0, stream>>>(z, wot, bO, out);
}
